// LinearClassifier_75084618268737
// MI455X (gfx1250) — matrix operand order repaired and checked by reading, not compiled, not run on hardware
//
#include <hip/hip_runtime.h>
#include <cstdint>
#include <cstddef>

// ---------------------------------------------------------------------------
// MI455X (gfx1250) fused VQ-classifier evaluation.
//   - bf16 WMMA (v_wmma_f32_16x16x32_bf16), f32 accumulate
//   - B tiles double-buffered in LDS via global_load_async_to_lds_b128
//     (ASYNCcnt), shared by all 16 waves of the block
//   - fused streaming reductions: never materialize [16384 x 8192] matrices
// ---------------------------------------------------------------------------

typedef __attribute__((ext_vector_type(16))) __bf16 v16bf;
typedef __attribute__((ext_vector_type(8)))  __bf16 v8bf;
typedef __attribute__((ext_vector_type(8)))  float  v8f;

#define DIMD   512
#define KEMB   8192
#define BSZ    16384

#define ROW_B  1040                 // 1024 B of data + 16 B pad (bank skew 4c)
#define TILE_B (16 * ROW_B)         // 16,640 B per B tile
#define NTHR   512                  // 16 waves per block

// --- async staging: one 16x512 bf16 tile of W into LDS buffer (nt&1) -------
// 512 threads x 2 chunks x 16 B = 16 KB.  2 async loads per wave per tile.
__device__ __forceinline__ void stage_issue(const __bf16* __restrict__ W, int nt,
                                            char* smem, int tid) {
  char* buf = smem + (nt & 1) * TILE_B;
#pragma unroll
  for (int j = 0; j < 2; ++j) {
    int e   = tid + j * NTHR;                 // 16-byte chunk id, 0..1023
    int row = e >> 6;                         // 64 chunks per 1024 B row
    int col = e & 63;
    const char* g = (const char*)W + (size_t)(nt * 16 + row) * 1024 + col * 16;
    uint32_t l = (uint32_t)(uintptr_t)(buf + row * ROW_B + col * 16);
    asm volatile("global_load_async_to_lds_b128 %0, %1, off"
                 :: "v"(l), "v"(g) : "memory");
  }
}
__device__ __forceinline__ void wait_async_le2() {
  asm volatile("s_wait_asynccnt 2" ::: "memory");
}
__device__ __forceinline__ void wait_async_0() {
  asm volatile("s_wait_asynccnt 0" ::: "memory");
}

// --- preload A m-tile (16 x 512 bf16) into 128 VGPRs ------------------------
__device__ __forceinline__ void load_A(const __bf16* __restrict__ Abase, v16bf a[16]) {
#pragma unroll
  for (int kt = 0; kt < 16; ++kt) {
    v8bf alo = *(const v8bf*)(Abase + kt * 32);       // K = kt*32 + g*8
    v8bf ahi = *(const v8bf*)(Abase + kt * 32 + 16);  // K = kt*32 + 16 + g*8
    a[kt] = __builtin_shufflevector(alo, ahi, 0,1,2,3,4,5,6,7,8,9,10,11,12,13,14,15);
  }
}

// --- one 16x16 output tile: A(regs) x B(LDS tile) ---------------------------
// rowbase = buf + c*ROW_B + g*16  (lane's row of W, k = kt*32+g*8+s, kt*32+16+g*8+s)
__device__ __forceinline__ v8f gemm_lds(const v16bf a[16], const char* rowbase) {
  v8f acc = {};
#pragma unroll
  for (int kt = 0; kt < 16; ++kt) {
    v8bf blo = *(const v8bf*)(rowbase + kt * 64);
    v8bf bhi = *(const v8bf*)(rowbase + kt * 64 + 32);
    v16bf b = __builtin_shufflevector(blo, bhi, 0,1,2,3,4,5,6,7,8,9,10,11,12,13,14,15);
    acc = __builtin_amdgcn_wmma_f32_16x16x32_bf16(false, a[kt], false, b,
                                                  (short)0, acc, false, false);
  }
  return acc;
}

// --- f32 -> bf16 elementwise convert ----------------------------------------
__global__ void k_cvt_bf16(const float* __restrict__ s, __bf16* __restrict__ d, int n) {
  int i = blockIdx.x * blockDim.x + threadIdx.x;
  if (i < n) d[i] = (__bf16)s[i];
}

// --- per-row squared norms of embeddings (exact f32) ------------------------
__global__ __launch_bounds__(256) void k_sqnorm(const float* __restrict__ E,
                                                float* __restrict__ sqn) {
  int lane = threadIdx.x & 31;
  int row  = blockIdx.x * 8 + (threadIdx.x >> 5);
  const float* p = E + (size_t)row * DIMD;
  float s = 0.f;
#pragma unroll
  for (int i = 0; i < DIMD / 32; ++i) { float v = p[i * 32 + lane]; s += v * v; }
#pragma unroll
  for (int off = 16; off > 0; off >>= 1) s += __shfl_xor(s, off, 32);
  if (lane == 0) sqn[row] = s;
}

// --- phase 1: nearest-codebook indices (streaming argmin over K=8192) -------
__global__ __launch_bounds__(NTHR) void k_argmin(const __bf16* __restrict__ melb,
                                                 const __bf16* __restrict__ embb,
                                                 const float*  __restrict__ sqn,
                                                 int* __restrict__ melidx) {
  __shared__ char smem[2 * TILE_B];
  int tid = threadIdx.x, lane = tid & 31, wave = tid >> 5;
  int g = lane >> 4, c = lane & 15;
  int m0 = (blockIdx.x * 16 + wave) * 16;

  v16bf a[16];
  load_A(melb + (size_t)(m0 + c) * DIMD + g * 8, a);

  float minv[8]; int mini[8];
#pragma unroll
  for (int v = 0; v < 8; ++v) { minv[v] = __builtin_inff(); mini[v] = 0; }

  stage_issue(embb, 0, smem, tid);
  const int NT = KEMB / 16;
  for (int nt = 0; nt < NT; ++nt) {
    if (nt + 1 < NT) { stage_issue(embb, nt + 1, smem, tid); wait_async_le2(); }
    else             { wait_async_0(); }
    __syncthreads();
    const char* buf = smem + (nt & 1) * TILE_B;
    v8f acc = gemm_lds(a, buf + c * ROW_B + g * 16);
    int n = nt * 16 + c;
    float bias = sqn[n];
#pragma unroll
    for (int v = 0; v < 8; ++v) {
      float dist = bias - 2.0f * acc[v];
      if (dist < minv[v]) { minv[v] = dist; mini[v] = n; }   // first-min kept
    }
    __syncthreads();
  }
#pragma unroll
  for (int off = 1; off < 16; off <<= 1) {
#pragma unroll
    for (int v = 0; v < 8; ++v) {
      float ov = __shfl_xor(minv[v], off, 32);
      int   oi = __shfl_xor(mini[v], off, 32);
      if (ov < minv[v] || (ov == minv[v] && oi < mini[v])) { minv[v] = ov; mini[v] = oi; }
    }
  }
  if (c == 0) {
#pragma unroll
    for (int v = 0; v < 8; ++v) melidx[m0 + g * 8 + v] = mini[v];
  }
}

// --- phase 2: h = LeakyReLU(eeg @ W1^T + b1) -> bf16 ------------------------
__global__ __launch_bounds__(NTHR) void k_mlp1(const __bf16* __restrict__ eegb,
                                               const __bf16* __restrict__ w1b,
                                               const float*  __restrict__ b1,
                                               __bf16* __restrict__ hb) {
  __shared__ char smem[2 * TILE_B];
  int tid = threadIdx.x, lane = tid & 31, wave = tid >> 5;
  int g = lane >> 4, c = lane & 15;
  int m0 = (blockIdx.x * 16 + wave) * 16;

  v16bf a[16];
  load_A(eegb + (size_t)(m0 + c) * DIMD + g * 8, a);

  stage_issue(w1b, 0, smem, tid);
  const int NT = DIMD / 16;
  for (int nt = 0; nt < NT; ++nt) {
    if (nt + 1 < NT) { stage_issue(w1b, nt + 1, smem, tid); wait_async_le2(); }
    else             { wait_async_0(); }
    __syncthreads();
    const char* buf = smem + (nt & 1) * TILE_B;
    v8f acc = gemm_lds(a, buf + c * ROW_B + g * 16);
    int n = nt * 16 + c;
    float bias = b1[n];
#pragma unroll
    for (int v = 0; v < 8; ++v) {
      float h = acc[v] + bias;
      h = (h >= 0.f) ? h : 0.01f * h;                        // LeakyReLU
      hb[(size_t)(m0 + g * 8 + v) * DIMD + n] = (__bf16)h;
    }
    __syncthreads();
  }
}

// --- phase 3: logits = h @ W2^T + b2; fused argmax + online LSE + gather ----
__global__ __launch_bounds__(NTHR) void k_logits(const __bf16* __restrict__ hb,
                                                 const __bf16* __restrict__ w2b,
                                                 const float*  __restrict__ b2,
                                                 const int*    __restrict__ melidx,
                                                 float* __restrict__ rowloss,
                                                 float* __restrict__ rowcorr) {
  __shared__ char smem[2 * TILE_B];
  int tid = threadIdx.x, lane = tid & 31, wave = tid >> 5;
  int g = lane >> 4, c = lane & 15;
  int m0 = (blockIdx.x * 16 + wave) * 16;

  v16bf a[16];
  load_A(hb + (size_t)(m0 + c) * DIMD + g * 8, a);

  int   midx[8];
  float mx[8], sm[8], amv[8], mv[8]; int ami[8];
#pragma unroll
  for (int v = 0; v < 8; ++v) {
    midx[v] = melidx[m0 + g * 8 + v];
    mx[v] = -__builtin_inff(); sm[v] = 0.f;
    amv[v] = -__builtin_inff(); ami[v] = 0; mv[v] = 0.f;
  }

  stage_issue(w2b, 0, smem, tid);
  const int NT = KEMB / 16;
  for (int nt = 0; nt < NT; ++nt) {
    if (nt + 1 < NT) { stage_issue(w2b, nt + 1, smem, tid); wait_async_le2(); }
    else             { wait_async_0(); }
    __syncthreads();
    const char* buf = smem + (nt & 1) * TILE_B;
    v8f acc = gemm_lds(a, buf + c * ROW_B + g * 16);
    int n = nt * 16 + c;
    float bias = b2[n];
#pragma unroll
    for (int v = 0; v < 8; ++v) {
      float l = acc[v] + bias;
      if (l > amv[v]) { amv[v] = l; ami[v] = n; }            // running argmax
      float nm = fmaxf(mx[v], l);                            // online log-sum-exp
      sm[v] = sm[v] * __expf(mx[v] - nm) + __expf(l - nm);
      mx[v] = nm;
      if (n == midx[v]) mv[v] = l;                           // gather target logit
    }
    __syncthreads();
  }
#pragma unroll
  for (int off = 1; off < 16; off <<= 1) {
#pragma unroll
    for (int v = 0; v < 8; ++v) {
      float omx = __shfl_xor(mx[v],  off, 32);
      float osm = __shfl_xor(sm[v],  off, 32);
      float nm  = fmaxf(mx[v], omx);
      sm[v] = sm[v] * __expf(mx[v] - nm) + osm * __expf(omx - nm);
      mx[v] = nm;
      float oav = __shfl_xor(amv[v], off, 32);
      int   oai = __shfl_xor(ami[v], off, 32);
      if (oav > amv[v] || (oav == amv[v] && oai < ami[v])) { amv[v] = oav; ami[v] = oai; }
      mv[v] += __shfl_xor(mv[v], off, 32);                   // exactly one lane matched
    }
  }
  if (c == 0) {
#pragma unroll
    for (int v = 0; v < 8; ++v) {
      int r = m0 + g * 8 + v;
      rowloss[r] = (mx[v] + __logf(sm[v])) - mv[v];          // -log p[mel_idx]
      rowcorr[r] = (ami[v] == midx[v]) ? 1.f : 0.f;
    }
  }
}

// --- final mean-reduction to (loss, acc) ------------------------------------
__global__ __launch_bounds__(256) void k_final(const float* __restrict__ rowloss,
                                               const float* __restrict__ rowcorr,
                                               float* __restrict__ out) {
  __shared__ float sL[256], sC[256];
  int t = threadIdx.x;
  float L = 0.f, C = 0.f;
  for (int i = t; i < BSZ; i += 256) { L += rowloss[i]; C += rowcorr[i]; }
  sL[t] = L; sC[t] = C; __syncthreads();
  for (int s = 128; s > 0; s >>= 1) {
    if (t < s) { sL[t] += sL[t + s]; sC[t] += sC[t + s]; }
    __syncthreads();
  }
  if (t == 0) { out[0] = sL[0] / (float)BSZ; out[1] = sC[0] / (float)BSZ; }
}

// ---------------------------------------------------------------------------
extern "C" void kernel_launch(void* const* d_in, const int* in_sizes, int n_in,
                              void* d_out, int out_size, void* d_ws, size_t ws_size,
                              hipStream_t stream) {
  const float* eeg = (const float*)d_in[0];   // [16384, 512]
  const float* mel = (const float*)d_in[1];   // [16384, 512]
  const float* emb = (const float*)d_in[2];   // [8192, 512]
  const float* W1  = (const float*)d_in[3];   // [512, 512]
  const float* b1  = (const float*)d_in[4];   // [512]
  const float* W2  = (const float*)d_in[5];   // [8192, 512]
  const float* b2  = (const float*)d_in[6];   // [8192]
  float* out = (float*)d_out;                 // [loss, acc]

  char* ws = (char*)d_ws;
  const size_t N_EEG = (size_t)BSZ * DIMD;    // 8,388,608
  const size_t N_EMB = (size_t)KEMB * DIMD;   // 4,194,304
  const size_t N_W1  = (size_t)DIMD * DIMD;   // 262,144

  __bf16* eegB = (__bf16*)(ws);                              // 16 MB
  __bf16* melB = (__bf16*)(ws + 16777216);                   // 16 MB
  __bf16* embB = (__bf16*)(ws + 33554432);                   //  8 MB
  __bf16* w1B  = (__bf16*)(ws + 41943040);                   // 0.5 MB
  __bf16* w2B  = (__bf16*)(ws + 42467328);                   //  8 MB
  __bf16* hB   = (__bf16*)(ws + 50855936);                   // 16 MB
  float*  sqn  = (float*) (ws + 67633152);                   // 32 KB
  int*    midx = (int*)   (ws + 67665920);                   // 64 KB
  float*  rls  = (float*) (ws + 67731456);                   // 64 KB
  float*  rcr  = (float*) (ws + 67796992);                   // 64 KB

  // f32 -> bf16 staging
  k_cvt_bf16<<<(int)((N_EEG + 255) / 256), 256, 0, stream>>>(eeg, eegB, (int)N_EEG);
  k_cvt_bf16<<<(int)((N_EEG + 255) / 256), 256, 0, stream>>>(mel, melB, (int)N_EEG);
  k_cvt_bf16<<<(int)((N_EMB + 255) / 256), 256, 0, stream>>>(emb, embB, (int)N_EMB);
  k_cvt_bf16<<<(int)((N_W1  + 255) / 256), 256, 0, stream>>>(W1,  w1B,  (int)N_W1);
  k_cvt_bf16<<<(int)((N_EMB + 255) / 256), 256, 0, stream>>>(W2,  w2B,  (int)N_EMB);

  // ||e_k||^2 (exact f32)
  k_sqnorm<<<KEMB / 8, 256, 0, stream>>>(emb, sqn);

  // fused phases: 64 blocks x 16 waves, each wave owns one 16-row M-tile
  k_argmin<<<BSZ / 256, NTHR, 0, stream>>>(melB, embB, sqn, midx);
  k_mlp1  <<<BSZ / 256, NTHR, 0, stream>>>(eegB, w1B, b1, hB);
  k_logits<<<BSZ / 256, NTHR, 0, stream>>>(hB, w2B, b2, midx, rls, rcr);
  k_final <<<1, 256, 0, stream>>>(rls, rcr, out);
}
